// AFNB_23845658427901
// MI455X (gfx1250) — matrix operand order repaired and checked by reading, not compiled, not run on hardware
//
#include <hip/hip_runtime.h>

typedef __attribute__((ext_vector_type(16))) __bf16 v16bf;
typedef __attribute__((ext_vector_type(8)))  __bf16 v8bf;
typedef __attribute__((ext_vector_type(8)))  float  v8f;
typedef __attribute__((ext_vector_type(4)))  unsigned int v4u;
typedef __attribute__((ext_vector_type(8)))  int v8i;
typedef __attribute__((ext_vector_type(4)))  int v4i;

#define LSTR 48   // padded LDS row stride (bf16 elems): 64B data + 32B pad = 96B

__device__ __forceinline__ v16bf ld_frag(const __bf16* p0, const __bf16* p1) {
  union { v16bf v; v8bf h[2]; } u;
  u.h[0] = *(const v8bf*)p0;
  u.h[1] = *(const v8bf*)p1;
  return u.v;
}

__device__ __forceinline__ v8f wmma_bf16(v16bf a, v16bf b, v8f c) {
  return __builtin_amdgcn_wmma_f32_16x16x32_bf16(false, a, false, b, (short)0, c, false, false);
}

// ---------------------------------------------------------------------------
// TDM: DMA one [128 rows x 32 bf16] tile global->LDS, padding each 64B row
// with 32B so LDS row stride is exactly LSTR (48 elems / 96B).
// D# per CDNA5 ISA 8.3-8.6: count=1, type=2(image), data_size=1(2B),
// pad_enable, pad_interval=3 (16 DWORDs), pad_amount=7 (8 DWORDs),
// tile 32x128, tensor dims 32x128, dim0 stride = ld (elems). Groups 2/3 zero.
// ---------------------------------------------------------------------------
__device__ __forceinline__ void tdm_load_tile(const void* gsrc, unsigned lds_off,
                                              int ld_elems) {
  const unsigned long long ga = (unsigned long long)(size_t)gsrc;
  v4u g0;
  g0[0] = 1u;                                               // count=1 (valid)
  g0[1] = lds_off;                                          // lds_addr (bytes)
  g0[2] = (unsigned)ga;                                     // global_addr[31:0]
  g0[3] = ((unsigned)(ga >> 32) & 0x01FFFFFFu) | (2u << 30);// addr[56:32] | type=2
  v8i g1;
  g1[0] = (int)((1u << 16) | (1u << 20) | (3u << 22) | (7u << 25));
  g1[1] = (int)(32u << 16);     // tensor_dim0 = 32
  g1[2] = (int)(128u << 16);    // tensor_dim1 = 128
  g1[3] = (int)(32u << 16);     // tile_dim0 = 32
  g1[4] = 128;                  // tile_dim1 = 128
  g1[5] = ld_elems;             // tensor_dim0_stride (elems)
  g1[6] = 0;
  g1[7] = 0;
  v4i z4 = {0, 0, 0, 0};
#if __has_include(<hip/amd_detail/amd_gfx1250_TDM.h>)
  v8i z8 = {0, 0, 0, 0, 0, 0, 0, 0};
  __builtin_amdgcn_tensor_load_to_lds(g0, g1, z4, z4, z8, 0);
#else
  __builtin_amdgcn_tensor_load_to_lds(g0, g1, z4, z4, 0);
#endif
}

// ---------------------------------------------------------------------------
// f32 -> bf16 strided copy-convert: dst[r*dld + c] = bf16(src[r*sld + soff + c])
// ---------------------------------------------------------------------------
__global__ void cvt_strided(const float* __restrict__ src, int sld, int soff,
                            __bf16* __restrict__ dst, int dld, int R, int C) {
  long total = (long)R * C;
  for (long i = (long)blockIdx.x * blockDim.x + threadIdx.x; i < total;
       i += (long)gridDim.x * blockDim.x) {
    long r = i / C, c = i - r * C;
    dst[r * dld + c] = (__bf16)src[r * sld + soff + c];
  }
}

// ---------------------------------------------------------------------------
// 32x32 tiled transpose + f32->bf16: dst[j1][j0] = src[j0][j1]
// ---------------------------------------------------------------------------
__global__ void transpose_cvt(const float* __restrict__ src, long src_bstride,
                              __bf16* __restrict__ dst, long dst_bstride,
                              int dstride, int S0, int S1) {
  __shared__ float tile[32][33];
  const int b  = blockIdx.z;
  const int j1 = blockIdx.x * 32;
  const int j0 = blockIdx.y * 32;
  const float* s = src + (long)b * src_bstride;
  __bf16*      d = dst + (long)b * dst_bstride;
  const int tx = threadIdx.x, ty = threadIdx.y;   // 32 x 8
#pragma unroll
  for (int i = 0; i < 32; i += 8)
    tile[ty + i][tx] = s[(long)(j0 + ty + i) * S1 + (j1 + tx)];
  __syncthreads();
#pragma unroll
  for (int i = 0; i < 32; i += 8)
    d[(long)(j1 + ty + i) * dstride + (j0 + tx)] = (__bf16)tile[tx][ty + i];
}

// ---------------------------------------------------------------------------
// b_eff[o] = dot(bn_w[o, 0:2048], o_b)
// ---------------------------------------------------------------------------
__global__ void beff_kernel(const float* __restrict__ bn_w,
                            const float* __restrict__ o_b,
                            float* __restrict__ beff) {
  __shared__ float red[256];
  const int o = blockIdx.x, t = threadIdx.x;
  float s = 0.f;
  for (int j = t; j < 2048; j += 256) s += bn_w[(long)o * 4096 + j] * o_b[j];
  red[t] = s; __syncthreads();
  for (int k = 128; k; k >>= 1) { if (t < k) red[t] += red[t + k]; __syncthreads(); }
  if (t == 0) beff[o] = red[0];
}

// ---------------------------------------------------------------------------
// Pyramid pooling (scales 1,3,6,8 -> 110 bins) of full [B][256][4096] maps.
// mode 0: k -> kpool_t [B][112][256] (rows 110..111 zero)
// mode 1: v -> vpool_t [B][256][128] (cols 110..127 zero)
// ---------------------------------------------------------------------------
__global__ void pool_kernel(const float* __restrict__ full,
                            __bf16* __restrict__ dst, int mode) {
  __shared__ float row[4096];
  const int c = blockIdx.x, b = blockIdx.y, t = threadIdx.x;
  const float* src = full + ((long)b * 256 + c) * 4096;
  for (int i = t; i < 4096; i += 256) row[i] = src[i];
  __syncthreads();
  const int lim = (mode == 0) ? 112 : 128;
  if (t < lim) {
    float val = 0.f;
    if (t < 110) {
      int s, r;
      if (t == 0)      { s = 1; r = 0; }
      else if (t < 10) { s = 3; r = t - 1; }
      else if (t < 46) { s = 6; r = t - 10; }
      else             { s = 8; r = t - 46; }
      const int i = r / s, j = r % s;
      const int y0 = (i * 64) / s, y1 = ((i + 1) * 64 + s - 1) / s;
      const int x0 = (j * 64) / s, x1 = ((j + 1) * 64 + s - 1) / s;
      float acc = 0.f;
      for (int y = y0; y < y1; ++y)
        for (int x = x0; x < x1; ++x) acc += row[y * 64 + x];
      val = acc / (float)((y1 - y0) * (x1 - x0));
    }
    if (mode == 0) dst[((long)b * 112 + t) * 256 + c] = (__bf16)val;
    else           dst[((long)b * 256 + c) * 128 + t] = (__bf16)val;
  }
}

// ---------------------------------------------------------------------------
// Generic NT WMMA GEMM: C[co][n] = A[co][:CI] . Bt[n][:CI]   (both bf16, K-contig)
// 256 thr (8 waves), tile 128co x 128n, K step 32. Tiles staged by the Tensor
// Data Mover (double-buffered LDS, TENSORcnt + barrier handoff); waves only do
// ds_load_b128 fragment reads + v_wmma.
// EPI: 0=q (BN+ReLU -> bf16 [n][256])    1=k (BN+ReLU -> f32 [co][n])
//      2=v (+bias   -> f32 [co][n])      3=out ((acc+beff)*BN -> f32 [co][n])
//      4=weff (bf16 scatter -> W_cat[co][2048+n])
// ---------------------------------------------------------------------------
template <int EPI>
__global__ __launch_bounds__(256)
void gemm_nt(const __bf16* __restrict__ A,
             const __bf16* __restrict__ Bt, long b_bstride,
             int CI, int ldb,
             const float* __restrict__ p_g, const float* __restrict__ p_b,
             const float* __restrict__ p_m, const float* __restrict__ p_v,
             const float* __restrict__ p_bias,
             float* __restrict__ outF, long outF_bstride, int ldn,
             __bf16* __restrict__ outB, long outB_bstride, int ldob, int oboff) {
  __shared__ __bf16 lsA[2][128 * LSTR];
  __shared__ __bf16 lsB[2][128 * LSTR];
  __shared__ float s_inv[128], s_bet[128];

  const int b   = blockIdx.z;
  const int n0  = blockIdx.x * 128;
  const int co0 = blockIdx.y * 128;
  const int t = threadIdx.x, lane = t & 31, wid = t >> 5;
  const int wco = (wid & 1) * 64;       // wave co offset inside block tile
  const int wn  = (wid >> 1) * 32;      // wave n  offset inside block tile
  const __bf16* Bb = Bt + (long)b * b_bstride;

  if (EPI == 0 || EPI == 1 || EPI == 3) {
    if (t < 128) {
      const int c = co0 + t;
      const float iv = p_g[c] * rsqrtf(p_v[c] + 1e-5f);
      s_inv[t] = iv;
      s_bet[t] = p_b[c] - p_m[c] * iv;
    }
  }

  v8f acc[4][2] = {};

  const int al  = lane & 15;
  const int ak8 = (lane & 16) ? 8 : 0;
  const int bk  = (lane & 16) ? 16 : 0;

  // TDM-stage k-step 0 into buffer 0
  if (wid == 0) {
    tdm_load_tile(A  + (long)co0 * CI,  (unsigned)(size_t)&lsA[0][0], CI);
    tdm_load_tile(Bb + (long)n0  * ldb, (unsigned)(size_t)&lsB[0][0], ldb);
    __builtin_amdgcn_s_wait_tensorcnt(0);
  }
  __syncthreads();

  const int nk = CI >> 5;
  for (int kk = 0; kk < nk; ++kk) {
    const int cur = kk & 1;
    if (kk + 1 < nk) {
      if (wid == 0) {  // DMA next tiles into the other buffer while we compute
        const int k0 = (kk + 1) << 5;
        tdm_load_tile(A  + (long)co0 * CI  + k0, (unsigned)(size_t)&lsA[cur ^ 1][0], CI);
        tdm_load_tile(Bb + (long)n0  * ldb + k0, (unsigned)(size_t)&lsB[cur ^ 1][0], ldb);
      }
      if (kk + 2 < nk) {  // warm L2 ahead of the DMA (global_prefetch_b8)
        const int kp = (kk + 2) << 5;
        __builtin_prefetch(A  + (long)(co0 + (t >> 1)) * CI  + kp, 0, 1);
        __builtin_prefetch(Bb + (long)(n0  + (t >> 1)) * ldb + kp, 0, 1);
      }
    }
    const __bf16* la = lsA[cur];
    const __bf16* lb = lsB[cur];
    v16bf af[4], bfv[2];
#pragma unroll
    for (int i = 0; i < 4; ++i) {
      const __bf16* p = la + (wco + i * 16 + al) * LSTR + ak8;
      af[i] = ld_frag(p, p + 16);
    }
#pragma unroll
    for (int j = 0; j < 2; ++j) {
      const __bf16* p = lb + (wn + j * 16 + al) * LSTR + ak8;
      bfv[j] = ld_frag(p, p + 16);
    }
#pragma unroll
    for (int i = 0; i < 4; ++i)
#pragma unroll
      for (int j = 0; j < 2; ++j)
        acc[i][j] = wmma_bf16(af[i], bfv[j], acc[i][j]);
    if (wid == 0) __builtin_amdgcn_s_wait_tensorcnt(0);
    __syncthreads();
  }

  // ---- epilogues ----
  const int hf8 = (lane & 16) ? 8 : 0;
#pragma unroll
  for (int i = 0; i < 4; ++i) {
#pragma unroll
    for (int j = 0; j < 2; ++j) {
      const int coL = wco + i * 16 + hf8;   // local co of VGPR r=0 (r adds 0..7)
      const int co  = co0 + coL;
      const int n   = n0 + wn + j * 16 + al;
      if (EPI == 0) {            // q: BN+ReLU, packed bf16 store into q_t[n][co..co+7]
        v8bf o;
#pragma unroll
        for (int r = 0; r < 8; ++r) {
          float y = acc[i][j][r] * s_inv[coL + r] + s_bet[coL + r];
          o[r] = (__bf16)fmaxf(y, 0.f);
        }
        *(v8bf*)(outB + (long)b * outB_bstride + (long)n * ldob + oboff + co) = o;
      } else if (EPI == 1) {     // k: BN+ReLU -> f32 [co][n]
#pragma unroll
        for (int r = 0; r < 8; ++r) {
          float y = acc[i][j][r] * s_inv[coL + r] + s_bet[coL + r];
          outF[(long)b * outF_bstride + (long)(co + r) * ldn + n] = fmaxf(y, 0.f);
        }
      } else if (EPI == 2) {     // v: +bias -> f32 [co][n]
#pragma unroll
        for (int r = 0; r < 8; ++r)
          outF[(long)b * outF_bstride + (long)(co + r) * ldn + n] =
              acc[i][j][r] + p_bias[co + r];
      } else if (EPI == 3) {     // out: (acc + b_eff)*inv + beta -> f32 [co][n]
#pragma unroll
        for (int r = 0; r < 8; ++r) {
          float y = (acc[i][j][r] + p_bias[co + r]) * s_inv[coL + r] + s_bet[coL + r];
          outF[(long)b * outF_bstride + (long)(co + r) * ldn + n] = y;
        }
      } else {                   // weff: bf16 -> W_cat[co][2048 + n]
#pragma unroll
        for (int r = 0; r < 8; ++r)
          outB[(long)(co + r) * ldob + oboff + n] = (__bf16)acc[i][j][r];
      }
    }
  }
}

// ---------------------------------------------------------------------------
// Attention: per wave 16 pixels. sim = q@k * 1/16 (7x 16x16 m-tiles, K=256),
// exact masked softmax over m=110 (16-lane shfl reductions), ctx = v^T @ sim^T
// (16 c-tiles, K=128 zero-padded), packed bf16 stores into X_cat cols 2048..2303.
// ---------------------------------------------------------------------------
__global__ __launch_bounds__(256)
void attn_kernel(const __bf16* __restrict__ qt,   // [B][4096][256]
                 const __bf16* __restrict__ kp,   // [B][112][256]
                 const __bf16* __restrict__ vp,   // [B][256][128]
                 __bf16* __restrict__ hc) {       // [B][4096][2304]
  __shared__ __bf16 lsim[8][16 * 136];
  const int b = blockIdx.y;
  const int t = threadIdx.x, lane = t & 31, wid = t >> 5;
  const int n0  = blockIdx.x * 128 + wid * 16;
  const int al  = lane & 15;
  const int ak8 = (lane & 16) ? 8 : 0;
  const int bk  = (lane & 16) ? 16 : 0;
  const int hf8 = (lane & 16) ? 8 : 0;

  const __bf16* qb = qt + ((long)b * 4096 + n0) * 256;
  const __bf16* kb = kp + (long)b * 112 * 256;
  const __bf16* vb = vp + (long)b * 256 * 128;

  v8f sim[7] = {};
#pragma unroll
  for (int kk = 0; kk < 8; ++kk) {
    const __bf16* qa = qb + (long)al * 256 + kk * 32 + ak8;
    const v16bf af = ld_frag(qa, qa + 16);
#pragma unroll
    for (int mt = 0; mt < 7; ++mt) {
      const __bf16* kq = kb + (long)(mt * 16 + al) * 256 + kk * 32 + ak8;
      sim[mt] = wmma_bf16(af, ld_frag(kq, kq + 16), sim[mt]);
    }
  }

  // masked softmax over m (per pixel row r: cols live in 16-lane half groups)
  float mx[8], sm[8];
#pragma unroll
  for (int r = 0; r < 8; ++r) mx[r] = -3.0e38f;
#pragma unroll
  for (int mt = 0; mt < 7; ++mt)
#pragma unroll
    for (int r = 0; r < 8; ++r) {
      float s = sim[mt][r] * 0.0625f;               // 256^-0.5
      if (mt == 6 && al >= 14) s = -3.0e38f;        // mask m >= 110
      sim[mt][r] = s;
      mx[r] = fmaxf(mx[r], s);
    }
#pragma unroll
  for (int r = 0; r < 8; ++r)
#pragma unroll
    for (int off = 1; off < 16; off <<= 1) mx[r] = fmaxf(mx[r], __shfl_xor(mx[r], off));
#pragma unroll
  for (int r = 0; r < 8; ++r) sm[r] = 0.f;
#pragma unroll
  for (int mt = 0; mt < 7; ++mt)
#pragma unroll
    for (int r = 0; r < 8; ++r) {
      const float p = __expf(sim[mt][r] - mx[r]);
      sim[mt][r] = p;
      sm[r] += p;
    }
#pragma unroll
  for (int r = 0; r < 8; ++r) {
#pragma unroll
    for (int off = 1; off < 16; off <<= 1) sm[r] += __shfl_xor(sm[r], off);
    sm[r] = 1.f / sm[r];
  }
  __bf16* ms = &lsim[wid][0];
#pragma unroll
  for (int mt = 0; mt < 7; ++mt)
#pragma unroll
    for (int r = 0; r < 8; ++r)
      ms[(r + hf8) * 136 + mt * 16 + al] = (__bf16)(sim[mt][r] * sm[r]);
#pragma unroll
  for (int r = 0; r < 8; ++r)             // zero K pad 112..127
    ms[(r + hf8) * 136 + 112 + al] = (__bf16)0.f;

  // ctx[c][n] = v^T (c x m) @ sim^T (m x n), K = 128
  v8f ct[16] = {};
#pragma unroll
  for (int k2 = 0; k2 < 4; ++k2) {
    const __bf16* sp = ms + al * 136 + k2 * 32 + ak8;
    const v16bf bf = ld_frag(sp, sp + 16);
#pragma unroll
    for (int c = 0; c < 16; ++c) {
      const __bf16* va = vb + (long)(c * 16 + al) * 128 + k2 * 32 + ak8;
      ct[c] = wmma_bf16(ld_frag(va, va + 16), bf, ct[c]);
    }
  }
  __bf16* outp = hc + ((long)b * 4096 + n0 + al) * 2304 + 2048;
#pragma unroll
  for (int c = 0; c < 16; ++c) {
    v8bf o;
#pragma unroll
    for (int r = 0; r < 8; ++r) o[r] = (__bf16)ct[c][r];
    *(v8bf*)(outp + c * 16 + hf8) = o;     // 8 consecutive channels -> b128 store
  }
}

// ---------------------------------------------------------------------------
extern "C" void kernel_launch(void* const* d_in, const int* in_sizes, int n_in,
                              void* d_out, int out_size, void* d_ws, size_t ws_size,
                              hipStream_t stream) {
  const float* low   = (const float*)d_in[0];
  const float* high  = (const float*)d_in[1];
  const float* q_w   = (const float*)d_in[2];
  const float* q_g   = (const float*)d_in[3];
  const float* q_bb  = (const float*)d_in[4];
  const float* q_m   = (const float*)d_in[5];
  const float* q_v   = (const float*)d_in[6];
  const float* k_w   = (const float*)d_in[7];
  const float* k_g   = (const float*)d_in[8];
  const float* k_bb  = (const float*)d_in[9];
  const float* k_m   = (const float*)d_in[10];
  const float* k_v   = (const float*)d_in[11];
  const float* v_w   = (const float*)d_in[12];
  const float* v_b   = (const float*)d_in[13];
  const float* o_w   = (const float*)d_in[14];
  const float* o_b   = (const float*)d_in[15];
  const float* bn_w  = (const float*)d_in[16];
  const float* bn_g  = (const float*)d_in[17];
  const float* bn_bb = (const float*)d_in[18];
  const float* bn_m  = (const float*)d_in[19];
  const float* bn_v  = (const float*)d_in[20];
  float* out = (float*)d_out;

  char* base = (char*)d_ws;
  size_t off = 0;
  auto carve = [&](size_t bytes) -> void* {
    void* r = base + off;
    off = (off + bytes + 255) & ~(size_t)255;
    return r;
  };
  __bf16* hc    = (__bf16*)carve(8ll * 4096 * 2304 * 2);  // X_cat: [high_bf16T | ctx]
  __bf16* lowt  = (__bf16*)carve(8ll * 4096 * 1024 * 2);
  __bf16* qt    = (__bf16*)carve(8ll * 4096 * 256 * 2);
  float*  kfull = (float*) carve(8ll * 256 * 4096 * 4);
  float*  vfull = (float*) carve(8ll * 256 * 4096 * 4);
  __bf16* kpool = (__bf16*)carve(8ll * 112 * 256 * 2);
  __bf16* vpool = (__bf16*)carve(8ll * 256 * 128 * 2);
  __bf16* qwb   = (__bf16*)carve(256ll * 2048 * 2);
  __bf16* kwb   = (__bf16*)carve(256ll * 1024 * 2);
  __bf16* vwb   = (__bf16*)carve(256ll * 1024 * 2);
  __bf16* bnwc  = (__bf16*)carve(2048ll * 2048 * 2);
  __bf16* wcat  = (__bf16*)carve(2048ll * 2304 * 2);      // [bn_w_high | W_eff]
  __bf16* owt   = (__bf16*)carve(256ll * 2048 * 2);
  float*  beff  = (float*) carve(2048ll * 4);

  // weight converts (f32 -> bf16, K-contiguous rows)
  cvt_strided<<<2048, 256, 0, stream>>>(q_w, 2048, 0, qwb, 2048, 256, 2048);
  cvt_strided<<<1024, 256, 0, stream>>>(k_w, 1024, 0, kwb, 1024, 256, 1024);
  cvt_strided<<<1024, 256, 0, stream>>>(v_w, 1024, 0, vwb, 1024, 256, 1024);
  cvt_strided<<<4096, 256, 0, stream>>>(bn_w, 4096, 0,    bnwc, 2048, 2048, 2048);
  cvt_strided<<<4096, 256, 0, stream>>>(bn_w, 4096, 2048, wcat, 2304, 2048, 2048);

  // activation transposes -> bf16 NT layout
  transpose_cvt<<<dim3(128, 64, 8), dim3(32, 8), 0, stream>>>(
      high, 2048ll * 4096, hc, 4096ll * 2304, 2304, 2048, 4096);
  transpose_cvt<<<dim3(128, 32, 8), dim3(32, 8), 0, stream>>>(
      low, 1024ll * 4096, lowt, 4096ll * 1024, 1024, 1024, 4096);
  transpose_cvt<<<dim3(8, 64, 1), dim3(32, 8), 0, stream>>>(
      o_w, 0, owt, 0, 2048, 2048, 256);

  beff_kernel<<<2048, 256, 0, stream>>>(bn_w, o_b, beff);

  // W_eff = bn_w[:, :2048] @ o_w  -> W_cat[:, 2048:2304]
  gemm_nt<4><<<dim3(2, 16, 1), 256, 0, stream>>>(
      bnwc, owt, 0, 2048, 2048, nullptr, nullptr, nullptr, nullptr, nullptr,
      nullptr, 0, 0, wcat, 0, 2304, 2048);

  // q = relu(BN(q_w @ high)) stored transposed bf16 [b][n][256]
  gemm_nt<0><<<dim3(32, 2, 8), 256, 0, stream>>>(
      qwb, hc, 4096ll * 2304, 2048, 2304, q_g, q_bb, q_m, q_v, nullptr,
      nullptr, 0, 0, qt, 4096ll * 256, 256, 0);

  // k = relu(BN(k_w @ low)) f32 [b][256][4096]
  gemm_nt<1><<<dim3(32, 2, 8), 256, 0, stream>>>(
      kwb, lowt, 4096ll * 1024, 1024, 1024, k_g, k_bb, k_m, k_v, nullptr,
      kfull, 256ll * 4096, 4096, nullptr, 0, 0, 0);

  // v = v_w @ low + v_b  f32 [b][256][4096]
  gemm_nt<2><<<dim3(32, 2, 8), 256, 0, stream>>>(
      vwb, lowt, 4096ll * 1024, 1024, 1024, nullptr, nullptr, nullptr, nullptr, v_b,
      vfull, 256ll * 4096, 4096, nullptr, 0, 0, 0);

  // pyramid pooling (deterministic)
  pool_kernel<<<dim3(256, 8), 256, 0, stream>>>(kfull, kpool, 0);
  pool_kernel<<<dim3(256, 8), 256, 0, stream>>>(vfull, vpool, 1);

  // attention -> ctx written into X_cat cols 2048..2303
  attn_kernel<<<dim3(32, 8), 256, 0, stream>>>(qt, kpool, vpool, hc);

  // out = BN(W_cat @ X_cat + b_eff)  f32 [b][2048][4096]
  gemm_nt<3><<<dim3(32, 16, 8), 256, 0, stream>>>(
      wcat, hc, 4096ll * 2304, 2304, 2304, bn_g, bn_bb, bn_m, bn_v, beff,
      out, 2048ll * 4096, 4096, nullptr, 0, 0, 0);
}
